// TFTv3_23931557773741
// MI455X (gfx1250) — matrix operand order repaired and checked by reading, not compiled, not run on hardware
//
#include <hip/hip_runtime.h>
#include <hip/hip_bf16.h>
#include <cstdint>

#define HDIM 128
#define FDIM 32
#define BSZ  2048
#define TLEN 168
#define PLEN 24
#define EPSV 1e-5f

typedef __attribute__((ext_vector_type(16))) _Float16 v16h;
typedef __attribute__((ext_vector_type(8)))  _Float16 v8h;
typedef __attribute__((ext_vector_type(8)))  float    v8f;

__device__ __forceinline__ float sigf(float x){ return 1.0f/(1.0f+__expf(-x)); }
__device__ __forceinline__ float eluf(float x){ return x>0.0f ? x : (__expf(x)-1.0f); }

__device__ __forceinline__ v16h cat8(v8h lo, v8h hi){
  return __builtin_shufflevector(lo,hi,0,1,2,3,4,5,6,7,8,9,10,11,12,13,14,15);
}

__device__ __forceinline__ v8f wmma16(v16h a, v16h b, v8f c){
  return __builtin_amdgcn_wmma_f32_16x16x32_f16(false, a, false, b, (short)0, c, false, false);
}

// A fragment (16xK, row-major LDS, row stride Krow halves), K-chunk k0.
__device__ __forceinline__ v16h load_a(const _Float16* act, int Krow, int k0, int lane){
  int m     = lane & 15;
  int hioff = (lane >> 4) << 4;           // 0 or 16
  const _Float16* p = act + m*Krow + (k0<<5) + hioff;
  v8h lo = *(const v8h*)p;
  v8h hi = *(const v8h*)(p + 8);
  return cat8(lo, hi);
}

// B fragment from LDS pre-swizzled so each lane reads 16 contiguous halves.
__device__ __forceinline__ v16h load_b(const _Float16* w, int KC, int n0, int k0, int lane){
  const _Float16* p = w + ((((n0*KC + k0) << 5) + lane) << 4);
  v8h lo = *(const v8h*)p;
  v8h hi = *(const v8h*)(p + 8);
  return cat8(lo, hi);
}

// Swizzle row-major [N][K] f32 weights into WMMA-B fragment order (f16, LDS).
// K / Ktot are compile-time powers of two -> pure shift/mask addressing.
template<int K, int Ktot>
__device__ __forceinline__ void stage_w(const float* __restrict__ src, _Float16* dst,
                                        int N, int kofs, int tid, int nth){
  constexpr int KC = Ktot >> 5;
  constexpr int KSH = (K == 32) ? 5 : 7;   // log2(K) for K in {32,128}
  for (int idx = tid; idx < N*K; idx += nth){
    int n = idx >> KSH, k = idx & (K - 1);
    float v = src[idx];
    int kk = k + kofs;
    int n0 = n >> 4, nl = n & 15, k0 = kk >> 5, kr = kk & 31;
    int lane = ((kr >> 4) << 4) + nl;
    int j = kr & 15;
    dst[((((n0*KC + k0) << 5) + lane) << 4) + j] = (_Float16)v;
  }
}

// Issue one async global->LDS DMA of a 16x128 f16 tile (16B per lane, 8 waves).
__device__ __forceinline__ void async_tile(const _Float16* __restrict__ src,
                                           _Float16* ldsdst, int t, int b0, int tid){
  int r  = tid >> 4;            // row 0..15
  int cb = (tid & 15) << 3;     // half column 0..120 step 8 (16B)
  const _Float16* gp = src + ((size_t)t*BSZ + (size_t)(b0 + r))*HDIM + cb;
  unsigned la = (unsigned)(uintptr_t)(ldsdst + r*128 + cb);
  asm volatile("global_load_async_to_lds_b128 %0, %1, off" :: "v"(la), "v"(gp) : "memory");
}

// ---------------------------------------------------------------------------
// Kernel 1: input projection  xp = elu(LN(x @ W_in.T + b_in)) -> f16 [T][B][H]
// ---------------------------------------------------------------------------
static constexpr unsigned PROJ_SMEM = 128*32*2 + 128*32*2 + 128*128*4 + 3*512;

__global__ void proj_kernel(const float* __restrict__ x, const float* __restrict__ Wi,
                            const float* __restrict__ bi, const float* __restrict__ g,
                            const float* __restrict__ be_, _Float16* __restrict__ X0)
{
  extern __shared__ char smem[];
  _Float16* wbuf = (_Float16*)smem;            // 128x32 swizzled
  _Float16* xa   = wbuf + 128*32;              // 128x32 activations
  char* q = (char*)(xa + 128*32);
  float* zb = (float*)q; q += 128*128*4;       // 128x128 f32
  float* bb = (float*)q; q += 512;
  float* gg = (float*)q; q += 512;
  float* bt = (float*)q; q += 512;

  int tid = threadIdx.x, nth = blockDim.x, lane = tid & 31, wave = tid >> 5;
  size_t r0 = (size_t)blockIdx.x * 128;

  stage_w<32,32>(Wi, wbuf, 128, 0, tid, nth);
  for (int i = tid; i < 128; i += nth){ bb[i] = bi[i]; gg[i] = g[i]; bt[i] = be_[i]; }
  for (int i = tid; i < 128*32; i += nth) xa[i] = (_Float16)x[r0*32 + i];
  __syncthreads();

  const _Float16* act = xa + wave*16*32;
  v16h afr = load_a(act, 32, 0, lane);
  int nlo = lane & 15, mhi = (lane >> 4) << 3;
  #pragma unroll
  for (int n0 = 0; n0 < 8; ++n0){
    v8f acc = {};
    acc = wmma16(afr, load_b(wbuf, 1, n0, 0, lane), acc);
    #pragma unroll
    for (int j = 0; j < 8; ++j)
      zb[(wave*16 + mhi + j)*128 + (n0 << 4) + nlo] = acc[j];
  }
  __syncthreads();

  if (tid < 128){
    int r = tid;
    float mu = 0.f;
    for (int h = 0; h < 128; ++h) mu += zb[r*128 + h] + bb[h];
    mu *= (1.f/128.f);
    float var = 0.f;
    for (int h = 0; h < 128; ++h){ float d = zb[r*128 + h] + bb[h] - mu; var += d*d; }
    var *= (1.f/128.f);
    float rs = rsqrtf(var + EPSV);
    size_t gr = r0 + r, b = gr / TLEN, t = gr % TLEN;
    _Float16* dstp = X0 + (t*(size_t)BSZ + b)*HDIM;
    for (int h = 0; h < 128; ++h){
      float v = (zb[r*128 + h] + bb[h] - mu)*rs*gg[h] + bt[h];
      dstp[h] = (_Float16)eluf(v);
    }
  }
}

// ---------------------------------------------------------------------------
// Kernel 2: full LSTM layer scan. Weights resident in LDS (256KB, swizzled).
// Issue-ahead-by-2 async DMA double buffer; only TWO barriers per step:
//   GEMM -> barrier A -> issue DMA(t+2) -> gates -> s_wait_asynccnt -> barrier B
// ---------------------------------------------------------------------------
static constexpr unsigned LSTM_SMEM =
  512*256*2 + 512*4 + 3*(16*128*2) + 16*512*4 + 16*128*4;   // 317440 B

__global__ void lstm_seq_kernel(const _Float16* __restrict__ src, _Float16* __restrict__ dst,
                                const float* __restrict__ Wih, const float* __restrict__ Whh,
                                const float* __restrict__ bih, const float* __restrict__ bhh,
                                float* __restrict__ hfin, float* __restrict__ cfin)
{
  extern __shared__ char smem[];
  _Float16* wb = (_Float16*)smem;                       // 512x256 swizzled
  char* q = smem + 512*256*2;
  float*    bias = (float*)q;    q += 512*4;
  _Float16* xs0  = (_Float16*)q; q += 16*128*2;         // x double buffer
  _Float16* xs1  = (_Float16*)q; q += 16*128*2;
  _Float16* hbuf = (_Float16*)q; q += 16*128*2;         // recurrent h (f16)
  float*    zb   = (float*)q;    q += 16*512*4;
  float*    cst  = (float*)q;    q += 16*128*4;

  int tid = threadIdx.x, nth = blockDim.x, lane = tid & 31, wave = tid >> 5;
  int b0 = blockIdx.x * 16;
  _Float16* xs[2] = { xs0, xs1 };

  stage_w<128,256>(Wih, wb, 512, 0,   tid, nth);
  stage_w<128,256>(Whh, wb, 512, 128, tid, nth);
  for (int i = tid; i < 512; i += nth) bias[i] = bih[i] + bhh[i];
  for (int i = tid; i < 16*128; i += nth){ cst[i] = 0.f; hbuf[i] = (_Float16)0.f; }

  async_tile(src, xs0, 0, b0, tid);                     // x(0) in flight
  if (TLEN > 1) async_tile(src, xs1, 1, b0, tid);       // x(1) in flight
  asm volatile("s_wait_asynccnt 1" ::: "memory");       // x(0) landed
  __syncthreads();

  int nlo = lane & 15, mhi = (lane >> 4) << 3;
  for (int t = 0; t < TLEN; ++t){
    // --- GEMM: z = [x_t | h] @ [Wih|Whh].T  (16x512, K=256) ---
    v16h afr[8];
    #pragma unroll
    for (int k0 = 0; k0 < 4; ++k0) afr[k0]     = load_a(xs[t & 1], 128, k0, lane);
    #pragma unroll
    for (int k0 = 0; k0 < 4; ++k0) afr[4 + k0] = load_a(hbuf,      128, k0, lane);

    for (int i = 0; i < 4; ++i){
      int n0 = wave*4 + i;
      v8f acc = {};
      #pragma unroll
      for (int k0 = 0; k0 < 8; ++k0)
        acc = wmma16(afr[k0], load_b(wb, 8, n0, k0, lane), acc);
      int n = (n0 << 4) + nlo;
      float bv = bias[n];
      #pragma unroll
      for (int j = 0; j < 8; ++j) zb[(mhi + j)*512 + n] = acc[j] + bv;
    }
    __syncthreads();                                    // barrier A

    // recycle slot (t&1): GEMM(t) is done reading it across all waves
    if (t + 2 < TLEN) async_tile(src, xs[t & 1], t + 2, b0, tid);

    // --- gates: z = [i|f|g|o]; 8 contiguous h per thread, vector stores ---
    {
      int base = tid << 3;
      int r = base >> 7, h0 = base & 127;
      const float* zr = zb + r*512;
      float* cr = cst + r*128 + h0;
      v8h hv;
      #pragma unroll
      for (int j = 0; j < 8; ++j){
        int h = h0 + j;
        float zi = zr[h], zf = zr[128 + h], zg = zr[256 + h], zo = zr[384 + h];
        float c2 = sigf(zf)*cr[j] + sigf(zi)*tanhf(zg);
        float h2 = sigf(zo)*tanhf(c2);
        cr[j] = c2;
        hv[j] = (_Float16)h2;
        if (t == TLEN - 1){
          hfin[(size_t)(b0 + r)*HDIM + h] = h2;
          cfin[(size_t)(b0 + r)*HDIM + h] = c2;
        }
      }
      *(v8h*)(hbuf + r*128 + h0) = hv;
      *(v8h*)(dst + ((size_t)t*BSZ + (b0 + r))*HDIM + h0) = hv;
    }

    if (t + 2 < TLEN)      asm volatile("s_wait_asynccnt 1" ::: "memory"); // x(t+1) landed
    else if (t + 1 < TLEN) asm volatile("s_wait_asynccnt 0" ::: "memory");
    __syncthreads();                                    // barrier B
  }
}

// ---------------------------------------------------------------------------
// Kernel 3: one decoder LSTM cell step.
// ---------------------------------------------------------------------------
static constexpr unsigned DEC_SMEM =
  512*256*2 + 512*4 + 16*256*2 + 16*512*4 + 512 + 512;             // 306176 B

__global__ void dec_lstm_kernel(const float* __restrict__ xin, const float* __restrict__ prev,
                                const float* __restrict__ decW, const float* __restrict__ decb,
                                const float* __restrict__ he, int mode,
                                const float* __restrict__ Wih, const float* __restrict__ Whh,
                                const float* __restrict__ bih, const float* __restrict__ bhh,
                                float* __restrict__ hstate, float* __restrict__ cstate)
{
  extern __shared__ char smem[];
  _Float16* wb = (_Float16*)smem;
  char* q = smem + 512*256*2;
  float*    bias = (float*)q;    q += 512*4;
  _Float16* xh   = (_Float16*)q; q += 16*256*2;
  float*    zb   = (float*)q;    q += 16*512*4;
  float*    cvec = (float*)q;    q += 512;
  float*    heb  = (float*)q;    q += 512;

  int tid = threadIdx.x, nth = blockDim.x, lane = tid & 31, wave = tid >> 5;
  int b0 = blockIdx.x * 16;

  stage_w<128,256>(Wih, wb, 512, 0,   tid, nth);
  stage_w<128,256>(Whh, wb, 512, 128, tid, nth);
  for (int i = tid; i < 512; i += nth) bias[i] = bih[i] + bhh[i];
  if (mode == 0){
    for (int i = tid; i < 128; i += nth) heb[i] = he[i];
    __syncthreads();
    for (int h = tid; h < 128; h += nth){
      float s = decb[h];
      for (int k = 0; k < 128; ++k) s += heb[k]*decW[h*129 + 1 + k];
      cvec[h] = s;
    }
  }
  __syncthreads();

  for (int i = tid; i < 16*128; i += nth){
    int r = i >> 7, h = i & 127, b = b0 + r;
    float xv = mode ? xin[(size_t)b*HDIM + h]
                    : (prev[b]*decW[h*129] + cvec[h]);
    xh[r*256 + h]       = (_Float16)xv;
    xh[r*256 + 128 + h] = (_Float16)hstate[(size_t)b*HDIM + h];
  }
  __syncthreads();

  v16h afr[8];
  #pragma unroll
  for (int k0 = 0; k0 < 8; ++k0) afr[k0] = load_a(xh, 256, k0, lane);
  int nlo = lane & 15, mhi = (lane >> 4) << 3;
  for (int i = 0; i < 4; ++i){
    int n0 = wave*4 + i;
    v8f acc = {};
    #pragma unroll
    for (int k0 = 0; k0 < 8; ++k0)
      acc = wmma16(afr[k0], load_b(wb, 8, n0, k0, lane), acc);
    int n = (n0 << 4) + nlo;
    float bv = bias[n];
    #pragma unroll
    for (int j = 0; j < 8; ++j) zb[(mhi + j)*512 + n] = acc[j] + bv;
  }
  __syncthreads();

  for (int i = tid; i < 16*128; i += nth){
    int r = i >> 7, h = i & 127, b = b0 + r;
    float zi = zb[r*512 + h];
    float zf = zb[r*512 + 128 + h];
    float zg = zb[r*512 + 256 + h];
    float zo = zb[r*512 + 384 + h];
    float c2 = sigf(zf)*cstate[(size_t)b*HDIM + h] + sigf(zi)*tanhf(zg);
    float h2 = sigf(zo)*tanhf(c2);
    cstate[(size_t)b*HDIM + h] = c2;
    hstate[(size_t)b*HDIM + h] = h2;
  }
}

// ---------------------------------------------------------------------------
// Kernel 4: GRN + LayerNorm + head + prev update for one decoder step.
// ---------------------------------------------------------------------------
static constexpr unsigned GRN_SMEM =
  3*128*128*2 + 2*(16*128*2) + 4*(16*128*4) + 5*512 + 3*128*4 + 16; // 143376 B

__global__ void grn_head_kernel(const float* __restrict__ xin,
                                const float* __restrict__ Wg, const float* __restrict__ bg,
                                const float* __restrict__ g,  const float* __restrict__ beta,
                                const float* __restrict__ headW, const float* __restrict__ headb,
                                float* __restrict__ prev, float* __restrict__ out, int p)
{
  extern __shared__ char smem[];
  _Float16* w0 = (_Float16*)smem;
  _Float16* w1 = w0 + 128*128;
  _Float16* w2 = w1 + 128*128;
  char* q = smem + 3*128*128*2;
  _Float16* xh  = (_Float16*)q; q += 16*128*2;
  _Float16* h1h = (_Float16*)q; q += 16*128*2;
  float* xf = (float*)q; q += 16*128*4;
  float* gb = (float*)q; q += 16*128*4;
  float* zb = (float*)q; q += 16*128*4;
  float* ob = (float*)q; q += 16*128*4;
  float* b0 = (float*)q; q += 512;
  float* b1 = (float*)q; q += 512;
  float* b2 = (float*)q; q += 512;
  float* gg = (float*)q; q += 512;
  float* be = (float*)q; q += 512;
  float* hw = (float*)q; q += 3*128*4;
  float* hb = (float*)q; q += 16;

  int tid = threadIdx.x, nth = blockDim.x, lane = tid & 31, wave = tid >> 5;
  int bb = blockIdx.x * 16;

  stage_w<128,128>(Wg,             w0, 128, 0, tid, nth);
  stage_w<128,128>(Wg +   128*128, w1, 128, 0, tid, nth);
  stage_w<128,128>(Wg + 2*128*128, w2, 128, 0, tid, nth);
  for (int i = tid; i < 128; i += nth){
    b0[i] = bg[i]; b1[i] = bg[128 + i]; b2[i] = bg[256 + i];
    gg[i] = g[i];  be[i] = beta[i];
  }
  for (int i = tid; i < 3*128; i += nth) hw[i] = headW[i];
  if (tid < 3) hb[tid] = headb[tid];
  for (int i = tid; i < 16*128; i += nth){
    float v = xin[(size_t)(bb + (i >> 7))*HDIM + (i & 127)];
    xf[i] = v; xh[i] = (_Float16)v;
  }
  __syncthreads();

  int nlo = lane & 15, mhi = (lane >> 4) << 3;
  {  // y0 = x@W0.T (-> elu -> h1h) and gate = sig(x@W2.T)
    v16h afr[4];
    #pragma unroll
    for (int k0 = 0; k0 < 4; ++k0) afr[k0] = load_a(xh, 128, k0, lane);
    int n0 = wave;
    v8f a0 = {}, a2 = {};
    #pragma unroll
    for (int k0 = 0; k0 < 4; ++k0){
      a0 = wmma16(afr[k0], load_b(w0, 4, n0, k0, lane), a0);
      a2 = wmma16(afr[k0], load_b(w2, 4, n0, k0, lane), a2);
    }
    int n = (n0 << 4) + nlo;
    #pragma unroll
    for (int j = 0; j < 8; ++j){
      int m = mhi + j;
      h1h[m*128 + n] = (_Float16)eluf(a0[j] + b0[n]);
      gb[m*128 + n]  = sigf(a2[j] + b2[n]);
    }
  }
  __syncthreads();
  {  // h2 = h1@W1.T + b1
    v16h afr[4];
    #pragma unroll
    for (int k0 = 0; k0 < 4; ++k0) afr[k0] = load_a(h1h, 128, k0, lane);
    int n0 = wave;
    v8f ac = {};
    #pragma unroll
    for (int k0 = 0; k0 < 4; ++k0)
      ac = wmma16(afr[k0], load_b(w1, 4, n0, k0, lane), ac);
    int n = (n0 << 4) + nlo;
    #pragma unroll
    for (int j = 0; j < 8; ++j) zb[(mhi + j)*128 + n] = ac[j] + b1[n];
  }
  __syncthreads();

  for (int i = tid; i < 16*128; i += nth) xf[i] = xf[i] + gb[i]*zb[i];
  __syncthreads();

  if (tid < 16){
    int r = tid;
    float mu = 0.f;
    for (int h = 0; h < 128; ++h) mu += xf[r*128 + h];
    mu *= (1.f/128.f);
    float var = 0.f;
    for (int h = 0; h < 128; ++h){ float d = xf[r*128 + h] - mu; var += d*d; }
    var *= (1.f/128.f);
    float rs = rsqrtf(var + EPSV);
    for (int h = 0; h < 128; ++h)
      ob[r*128 + h] = (xf[r*128 + h] - mu)*rs*gg[h] + be[h];
  }
  __syncthreads();

  if (tid < 48){
    int r = tid / 3, j = tid % 3;
    float s = hb[j];
    for (int h = 0; h < 128; ++h) s += ob[r*128 + h]*hw[j*128 + h];
    out[(size_t)j*BSZ*PLEN + (size_t)(bb + r)*PLEN + p] = s;
    if (j == 1) prev[bb + r] = s;
  }
}

__global__ void dec_init_kernel(const float* __restrict__ x, float* __restrict__ prev){
  int i = blockIdx.x*blockDim.x + threadIdx.x;
  if (i < BSZ) prev[i] = x[(size_t)i*TLEN*FDIM + (size_t)(TLEN - 1)*FDIM];
}

// ---------------------------------------------------------------------------
extern "C" void kernel_launch(void* const* d_in, const int* in_sizes, int n_in,
                              void* d_out, int out_size, void* d_ws, size_t ws_size,
                              hipStream_t stream)
{
  const float* x      = (const float*)d_in[0];
  const float* W_in   = (const float*)d_in[1];
  const float* b_in   = (const float*)d_in[2];
  const float* lng    = (const float*)d_in[3];
  const float* lnb    = (const float*)d_in[4];
  const float* encWih = (const float*)d_in[5];
  const float* encWhh = (const float*)d_in[6];
  const float* encbih = (const float*)d_in[7];
  const float* encbhh = (const float*)d_in[8];
  const float* decW   = (const float*)d_in[9];
  const float* decb   = (const float*)d_in[10];
  const float* decWih = (const float*)d_in[11];
  const float* decWhh = (const float*)d_in[12];
  const float* decbih = (const float*)d_in[13];
  const float* decbhh = (const float*)d_in[14];
  const float* dgrnW  = (const float*)d_in[15];
  const float* dgrnb  = (const float*)d_in[16];
  const float* dgrng  = (const float*)d_in[17];
  const float* dgrnbe = (const float*)d_in[18];
  const float* hourE  = (const float*)d_in[19];
  const float* headW  = (const float*)d_in[20];
  const float* headb  = (const float*)d_in[21];
  float* out = (float*)d_out;

  char* w = (char*)d_ws;
  size_t TBH = (size_t)TLEN*BSZ*HDIM;
  _Float16* X0 = (_Float16*)w;
  _Float16* X1 = (_Float16*)(w + TBH*sizeof(_Float16));
  char* st = w + 2*TBH*sizeof(_Float16);
  size_t SH = (size_t)BSZ*HDIM*sizeof(float);
  float* hD0  = (float*)(st + 0*SH);
  float* cD0  = (float*)(st + 1*SH);
  float* hD1  = (float*)(st + 2*SH);
  float* cD1  = (float*)(st + 3*SH);
  float* prev = (float*)(st + 4*SH);

  const int LW = 4*HDIM*HDIM;   // per-layer LSTM weight stride
  const int LB = 4*HDIM;        // per-layer bias stride

  proj_kernel<<<(BSZ*TLEN)/128, 256, PROJ_SMEM, stream>>>(x, W_in, b_in, lng, lnb, X0);

  lstm_seq_kernel<<<BSZ/16, 256, LSTM_SMEM, stream>>>(
      X0, X1, encWih, encWhh, encbih, encbhh, hD0, cD0);
  lstm_seq_kernel<<<BSZ/16, 256, LSTM_SMEM, stream>>>(
      X1, X0, encWih + LW, encWhh + LW, encbih + LB, encbhh + LB, hD1, cD1);

  dec_init_kernel<<<(BSZ + 255)/256, 256, 0, stream>>>(x, prev);

  for (int p = 0; p < PLEN; ++p){
    dec_lstm_kernel<<<BSZ/16, 256, DEC_SMEM, stream>>>(
        nullptr, prev, decW, decb, hourE + p*HDIM, 0,
        decWih, decWhh, decbih, decbhh, hD0, cD0);
    dec_lstm_kernel<<<BSZ/16, 256, DEC_SMEM, stream>>>(
        hD0, nullptr, decW, decb, nullptr, 1,
        decWih + LW, decWhh + LW, decbih + LB, decbhh + LB, hD1, cD1);
    grn_head_kernel<<<BSZ/16, 256, GRN_SMEM, stream>>>(
        hD1, dgrnW, dgrnb, dgrng, dgrnbe, headW, headb, prev, out, p);
  }
}
